// LSTM_43817256354365
// MI455X (gfx1250) — matrix operand order repaired and checked by reading, not compiled, not run on hardware
//
#include <hip/hip_runtime.h>
#include <hip/hip_bf16.h>

// ---------------------------------------------------------------------------
// LSTM cell on gfx1250 via V_WMMA_F32_16X16X32_BF16, software-pipelined,
// with uniform-base + 32-bit per-lane offset addressing (GVS form).
// B=4096, I=1024, H=2048, O=1024.  K3 = I+H = 3072 (fused [x|h] GEMM).
// ---------------------------------------------------------------------------

#define B_SZ  4096
#define I_SZ  1024
#define H_SZ  2048
#define O_SZ  1024
#define K3    (I_SZ + H_SZ)   // 3072

typedef __attribute__((ext_vector_type(16))) __bf16 bf16x16;
typedef __attribute__((ext_vector_type(8)))  float  f32x8;

union Frag {
    bf16x16 v;
    uint4   q[2];
};

__device__ __forceinline__ float sigmoidf_dev(float x) {
    return 1.0f / (1.0f + __expf(-x));
}

__device__ __forceinline__ uint4 ld128(const void* base, unsigned byteOff) {
    return *(const uint4*)((const char*)base + byteOff);
}

// ---------------------------------------------------------------------------
// Pack kernels: f32 -> bf16 staging buffers
// ---------------------------------------------------------------------------

// xh_bf[row][k] = k < I ? x[row][k] : h[row][k-I]     grid: (K3/256, B)
__global__ __launch_bounds__(256) void pack_xh_kernel(
    const float* __restrict__ x, const float* __restrict__ h,
    __bf16* __restrict__ out)
{
    const int k   = blockIdx.x * 256 + threadIdx.x;
    const int row = blockIdx.y;
    const float v = (k < I_SZ) ? x[(size_t)row * I_SZ + k]
                               : h[(size_t)row * H_SZ + (k - I_SZ)];
    out[(size_t)row * K3 + k] = (__bf16)v;
}

// W_bf[n][k], n in [0,4H): gate g=n/H, r=n%H ; k<I -> Wx_g[r][k] else Wh_g[r][k-I]
// grid: (K3/256, 4H)
__global__ __launch_bounds__(256) void pack_w_kernel(
    const float* __restrict__ Wxi, const float* __restrict__ Whi,
    const float* __restrict__ Wxf, const float* __restrict__ Whf,
    const float* __restrict__ Wxc, const float* __restrict__ Whc,
    const float* __restrict__ Wxo, const float* __restrict__ Who,
    __bf16* __restrict__ out)
{
    const int k   = blockIdx.x * 256 + threadIdx.x;
    const int row = blockIdx.y;             // 0..4H-1
    const int g   = row >> 11;              // row / H  (H = 2048)
    const int r   = row & (H_SZ - 1);
    const float* Wx = (g == 0) ? Wxi : (g == 1) ? Wxf : (g == 2) ? Wxc : Wxo;
    const float* Wh = (g == 0) ? Whi : (g == 1) ? Whf : (g == 2) ? Whc : Who;
    const float v = (k < I_SZ) ? Wx[(size_t)r * I_SZ + k]
                               : Wh[(size_t)r * H_SZ + (k - I_SZ)];
    out[(size_t)row * K3 + k] = (__bf16)v;
}

// Why_bf[row][k] = Why[row][k]   grid: (H/256, O)
__global__ __launch_bounds__(256) void pack_why_kernel(
    const float* __restrict__ Why, __bf16* __restrict__ out)
{
    const int k   = blockIdx.x * 256 + threadIdx.x;
    const int row = blockIdx.y;
    out[(size_t)row * H_SZ + k] = (__bf16)Why[(size_t)row * H_SZ + k];
}

// ---------------------------------------------------------------------------
// Fused gate GEMM + LSTM cell update.
// Block: 256 threads = 8 waves (wave32), tile BM=256 x BN=32 hidden columns.
// Wave: WM=64 (4 M-tiles) x WN=16, 4 gates -> 16 acc tiles (v8f each = 128 VGPR).
// Ping-pong double buffering; per-lane u32 byte offsets + uniform base so the
// K-loop advances with scalar adds instead of per-lane 64-bit VALU adds.
// ---------------------------------------------------------------------------
__global__ __launch_bounds__(256) void lstm_cell_kernel(
    const __bf16* __restrict__ A,      // [B][K3]
    const __bf16* __restrict__ W,      // [4H][K3] row-major
    const float*  __restrict__ bI,     // [H] per-gate biases
    const float*  __restrict__ bF,
    const float*  __restrict__ bC,
    const float*  __restrict__ bO,
    const float*  __restrict__ cell,   // [B][H]
    float*        __restrict__ outH,   // [B][H] new_hidden (f32)
    float*        __restrict__ outC,   // [B][H] new_cell   (f32)
    __bf16*       __restrict__ nh_bf)  // [B][H] new_hidden (bf16, for GEMM2)
{
    const int lane    = threadIdx.x & 31;
    const int wave    = threadIdx.x >> 5;     // 0..7
    const int waveM   = wave >> 1;            // 0..3
    const int waveN   = wave & 1;             // 0..1
    const int m0      = blockIdx.x * 256 + waveM * 64;
    const int j0      = blockIdx.y * 32  + waveN * 16;
    const int laneRow = lane & 15;
    const int hiHalf  = lane >> 4;            // 0 or 1

    const f32x8 zc = {};
    f32x8 acc[4][4];
#pragma unroll
    for (int g = 0; g < 4; ++g)
#pragma unroll
        for (int t = 0; t < 4; ++t)
            acc[g][t] = zc;

    // Per-lane, loop-invariant BYTE offsets (all < 2^31).
    // A fragment (16-bit 16x32 layout): lane<16 -> row M=lane, K 0..7 & 16..23
    //                                   lane>=16 -> row M=lane-16, K 8..15 & 24..31
    unsigned aOff[4];
#pragma unroll
    for (int t = 0; t < 4; ++t)
        aOff[t] = (unsigned)(m0 + t * 16 + laneRow) * (K3 * 2) + hiHalf * 16;
    // B fragment (32x16): lane<16 -> col N=lane, K 0..7 & 16..23 ; lane>=16 -> K 8..15 & 24..31.
    // Column n of B == row (j-index) of row-major W -> two runs of 8 contiguous bf16.
    unsigned wOff[4];
#pragma unroll
    for (int g = 0; g < 4; ++g)
        wOff[g] = (unsigned)(g * H_SZ + j0 + laneRow) * (K3 * 2) + hiHalf * 16;

    auto loadA = [&](Frag (&dst)[4], unsigned kkB) {
#pragma unroll
        for (int t = 0; t < 4; ++t) {
            dst[t].q[0] = ld128(A, aOff[t] + kkB);
            dst[t].q[1] = ld128(A, aOff[t] + kkB + 32);
        }
    };
    auto loadB = [&](Frag (&dst)[4], unsigned kkB) {
#pragma unroll
        for (int g = 0; g < 4; ++g) {
            dst[g].q[0] = ld128(W, wOff[g] + kkB);
            dst[g].q[1] = ld128(W, wOff[g] + kkB + 32);
        }
    };
    auto mma = [&](Frag (&a)[4], Frag (&b)[4]) {
#pragma unroll
        for (int g = 0; g < 4; ++g)
#pragma unroll
            for (int t = 0; t < 4; ++t)
                acc[g][t] = __builtin_amdgcn_wmma_f32_16x16x32_bf16(
                    false, a[t].v, false, b[g].v, (short)0, acc[g][t], false, false);
    };

    Frag aPing[4], bPing[4], aPong[4], bPong[4];
    loadA(aPing, 0);
    loadB(bPing, 0);
    // K3 = 3072 elements = 6144 bytes, step 128 bytes -> 48 iterations.
    for (unsigned kbB = 0; kbB < K3 * 2; kbB += 128) {
        loadA(aPong, kbB + 64);
        loadB(bPong, kbB + 64);
        mma(aPing, bPing);
        const unsigned knB = (kbB + 128 < K3 * 2) ? (kbB + 128) : 0;  // uniform clamp
        loadA(aPing, knB);
        loadB(bPing, knB);
        mma(aPong, bPong);
    }

    // Epilogue. C/D layout: VGPR r, lanes 0-15 -> M=r, N=lane; lanes 16-31 -> M=r+8.
    const int j = j0 + laneRow;
    const float bi = bI[j], bf = bF[j], bc = bC[j], bo = bO[j];
#pragma unroll
    for (int t = 0; t < 4; ++t) {
#pragma unroll
        for (int r = 0; r < 8; ++r) {
            const int row    = m0 + t * 16 + r + hiHalf * 8;
            const size_t idx = (size_t)row * H_SZ + j;
            const float gi = acc[0][t][r] + bi;
            const float gf = acc[1][t][r] + bf;
            const float gc = acc[2][t][r] + bc;
            const float go = acc[3][t][r] + bo;
            const float ig = sigmoidf_dev(gi);
            const float fg = sigmoidf_dev(gf);
            const float cn = tanhf(gc);
            const float og = sigmoidf_dev(go);
            const float cNew = fg * cell[idx] + ig * cn;
            const float hNew = og * tanhf(cNew);
            outC[idx]  = cNew;
            outH[idx]  = hNew;
            nh_bf[idx] = (__bf16)hNew;
        }
    }
}

// ---------------------------------------------------------------------------
// Output GEMM: out[B][O] = nh_bf @ Why_bf^T + bias.
// Same wave tiling (WM=64 x WN=16), ping-pong pipelined, K=H, u32 offsets.
// ---------------------------------------------------------------------------
__global__ __launch_bounds__(256) void out_gemm_kernel(
    const __bf16* __restrict__ A,     // [B][H]
    const __bf16* __restrict__ W,     // [O][H] row-major
    const float*  __restrict__ bias,  // [O]
    float*        __restrict__ out)   // [B][O]
{
    const int lane    = threadIdx.x & 31;
    const int wave    = threadIdx.x >> 5;
    const int waveM   = wave >> 1;
    const int waveN   = wave & 1;
    const int m0      = blockIdx.x * 256 + waveM * 64;
    const int j0      = blockIdx.y * 32  + waveN * 16;
    const int laneRow = lane & 15;
    const int hiHalf  = lane >> 4;

    const f32x8 zc = {};
    f32x8 acc[4];
#pragma unroll
    for (int t = 0; t < 4; ++t) acc[t] = zc;

    unsigned aOff[4];
#pragma unroll
    for (int t = 0; t < 4; ++t)
        aOff[t] = (unsigned)(m0 + t * 16 + laneRow) * (H_SZ * 2) + hiHalf * 16;
    const unsigned wOff = (unsigned)(j0 + laneRow) * (H_SZ * 2) + hiHalf * 16;

    auto loadA = [&](Frag (&dst)[4], unsigned kkB) {
#pragma unroll
        for (int t = 0; t < 4; ++t) {
            dst[t].q[0] = ld128(A, aOff[t] + kkB);
            dst[t].q[1] = ld128(A, aOff[t] + kkB + 32);
        }
    };
    auto loadB = [&](Frag& dst, unsigned kkB) {
        dst.q[0] = ld128(W, wOff + kkB);
        dst.q[1] = ld128(W, wOff + kkB + 32);
    };
    auto mma = [&](Frag (&a)[4], Frag& b) {
#pragma unroll
        for (int t = 0; t < 4; ++t)
            acc[t] = __builtin_amdgcn_wmma_f32_16x16x32_bf16(
                false, a[t].v, false, b.v, (short)0, acc[t], false, false);
    };

    Frag aPing[4], aPong[4];
    Frag bPing, bPong;
    loadA(aPing, 0);
    loadB(bPing, 0);
    for (unsigned kbB = 0; kbB < H_SZ * 2; kbB += 128) {
        loadA(aPong, kbB + 64);
        loadB(bPong, kbB + 64);
        mma(aPing, bPing);
        const unsigned knB = (kbB + 128 < H_SZ * 2) ? (kbB + 128) : 0;  // uniform clamp
        loadA(aPing, knB);
        loadB(bPing, knB);
        mma(aPong, bPong);
    }

    const int j = j0 + laneRow;
    const float bj = bias[j];
#pragma unroll
    for (int t = 0; t < 4; ++t) {
#pragma unroll
        for (int r = 0; r < 8; ++r) {
            const int row = m0 + t * 16 + r + hiHalf * 8;
            out[(size_t)row * O_SZ + j] = acc[t][r] + bj;
        }
    }
}

// ---------------------------------------------------------------------------
// Host launcher
// ---------------------------------------------------------------------------
extern "C" void kernel_launch(void* const* d_in, const int* in_sizes, int n_in,
                              void* d_out, int out_size, void* d_ws, size_t ws_size,
                              hipStream_t stream)
{
    (void)in_sizes; (void)n_in; (void)out_size; (void)ws_size;

    const float* x   = (const float*)d_in[0];
    const float* h   = (const float*)d_in[1];
    const float* c   = (const float*)d_in[2];
    const float* Wxi = (const float*)d_in[3];
    const float* bxi = (const float*)d_in[4];
    const float* Whi = (const float*)d_in[5];
    const float* Wxf = (const float*)d_in[6];
    const float* bxf = (const float*)d_in[7];
    const float* Whf = (const float*)d_in[8];
    const float* Wxc = (const float*)d_in[9];
    const float* bxc = (const float*)d_in[10];
    const float* Whc = (const float*)d_in[11];
    const float* Wxo = (const float*)d_in[12];
    const float* bxo = (const float*)d_in[13];
    const float* Who = (const float*)d_in[14];
    const float* Why = (const float*)d_in[15];
    const float* bhy = (const float*)d_in[16];

    float* out0 = (float*)d_out;                       // output     [B][O]
    float* outH = out0 + (size_t)B_SZ * O_SZ;          // new_hidden [B][H]
    float* outC = outH + (size_t)B_SZ * H_SZ;          // new_cell   [B][H]

    char* ws = (char*)d_ws;
    const size_t xh_bytes  = (size_t)B_SZ * K3 * 2;          // 24 MiB
    const size_t w_bytes   = (size_t)4 * H_SZ * K3 * 2;      // 48 MiB
    const size_t why_bytes = (size_t)O_SZ * H_SZ * 2;        // 4 MiB
    __bf16* xh_bf  = (__bf16*)(ws);
    __bf16* W_bf   = (__bf16*)(ws + xh_bytes);
    __bf16* Why_bf = (__bf16*)(ws + xh_bytes + w_bytes);
    __bf16* nh_bf  = (__bf16*)(ws + xh_bytes + w_bytes + why_bytes);

    // 1) pack inputs/weights to bf16
    pack_xh_kernel<<<dim3(K3 / 256, B_SZ), 256, 0, stream>>>(x, h, xh_bf);
    pack_w_kernel<<<dim3(K3 / 256, 4 * H_SZ), 256, 0, stream>>>(
        Wxi, Whi, Wxf, Whf, Wxc, Whc, Wxo, Who, W_bf);
    pack_why_kernel<<<dim3(H_SZ / 256, O_SZ), 256, 0, stream>>>(Why, Why_bf);

    // 2) fused gate GEMM + LSTM cell update  (grid 16 x 64, 8 waves/block)
    lstm_cell_kernel<<<dim3(B_SZ / 256, H_SZ / 32), 256, 0, stream>>>(
        xh_bf, W_bf, bxi, bxf, bxc, bxo, c, outH, outC, nh_bf);

    // 3) output projection  (grid 16 x 32)
    out_gemm_kernel<<<dim3(B_SZ / 256, O_SZ / 32), 256, 0, stream>>>(
        nh_bf, Why_bf, bhy, out0);
}
